// MinGRUForTimeSeries_32581621907475
// MI455X (gfx1250) — matrix operand order repaired and checked by reading, not compiled, not run on hardware
//
#include <hip/hip_runtime.h>

// MinGRU forward for B=16, L=4096, I=64, H=512, O=1 on gfx1250 (wave32, WMMA).
//
// Math: the reference's log-space parallel scan == linear recurrence
//   h_t = coeff_t * h_{t-1} + z_t * hw_t, with
//   k = sigmoid(x@Wz + bz); z = sigmoid(k); coeff = 1 - z;
//   hw = tanh(sigmoid(x@Wh + bh)) + 1e-8;  h_init = 1e-8 (= exp(log EPS)).
// All terms in (0,1) -> linear-space scan is stable. The scan is parallelized
// as a chunked first-order-recurrence scan: compose (P,Q) per chunk, combine
// chunk prefixes sequentially (C=64 steps), then re-apply per chunk.

typedef __attribute__((ext_vector_type(16))) _Float16 v16h;
typedef __attribute__((ext_vector_type(8)))  float    v8f;
typedef __attribute__((ext_vector_type(4)))  float    v4f;

#define GRU_EPS 1e-8f

__device__ __forceinline__ float fast_sigmoid(float x) {
    return 1.0f / (1.0f + __expf(-x));
}
__device__ __forceinline__ float fast_tanh(float x) {
    float e = __expf(-2.0f * x);
    return (1.0f - e) / (1.0f + e);
}

// ---------------------------------------------------------------------------
// Transpose + convert weights:  W[K,N] f32  ->  Wt[N,K] f16  (L2 resident)
// ---------------------------------------------------------------------------
__global__ void wt_convert_kernel(const float* __restrict__ W,
                                  _Float16* __restrict__ Wt,
                                  int K, int N) {
    int idx = blockIdx.x * blockDim.x + threadIdx.x;
    if (idx >= K * N) return;
    int k = idx / N;
    int n = idx - k * N;
    Wt[(size_t)n * K + k] = (_Float16)W[(size_t)k * N + n];
}

// A-fragment loader: row-major f32, converts to the WMMA f16 A lane layout.
// p points at X + rowA*K + khalf*16 (+ compile-time step offset).
__device__ __forceinline__ v16h load_a_frag(const float* __restrict__ p) {
    v4f x0 = *(const v4f*)(p + 0);
    v4f x1 = *(const v4f*)(p + 4);
    v4f x2 = *(const v4f*)(p + 8);
    v4f x3 = *(const v4f*)(p + 12);
    v16h a;
#pragma unroll
    for (int i = 0; i < 4; ++i) {
        a[i]      = (_Float16)x0[i];
        a[4 + i]  = (_Float16)x1[i];
        a[8 + i]  = (_Float16)x2[i];
        a[12 + i] = (_Float16)x3[i];
    }
    return a;
}

// ---------------------------------------------------------------------------
// Dual GEMM + fused activation epilogue. Templated on K so the K-loop fully
// unrolls: all loads use base-pointer + immediate offsets (no per-step u64
// math) and fragments are explicitly double-buffered so step s+1 loads are
// in flight while step s WMMAs execute.
//
// Block = 256 threads = 8 waves; wave (wm 0..3, wn 0..1) owns 16(M) x 64(N)
// for BOTH weight matrices (shared A fragment). Block tile 64x128.
// ---------------------------------------------------------------------------
template<int K>
__global__ __launch_bounds__(256) void gemm_act_kernel(
    const float* __restrict__ X,        // [M, K] f32
    const _Float16* __restrict__ WtZ,   // [N, K] f16 (transposed Wz)
    const _Float16* __restrict__ WtH,   // [N, K] f16 (transposed Wh)
    const float* __restrict__ bz,
    const float* __restrict__ bh,
    float* __restrict__ a_out,          // [M, N] coeff   (streaming / NT)
    float* __restrict__ b_out,          // [M, N] z*hw    (streaming / NT)
    int N)
{
    constexpr int NSTEP = K / 32;
    const int lane  = threadIdx.x & 31;
    const int wave  = threadIdx.x >> 5;
    const int wm    = wave & 3;
    const int wn    = wave >> 2;
    const int mRow  = blockIdx.x * 64 + wm * 16;
    const int nColW = blockIdx.y * 128 + wn * 64;

    const int rowA    = mRow + (lane & 15);
    const int khalf   = lane >> 4;
    const int colBase = nColW + (lane & 15);

    // Base pointers computed once; per-step addressing is immediate offsets.
    const float* pA = X + (size_t)rowA * K + khalf * 16;
    const _Float16* pZ[4];
    const _Float16* pH[4];
#pragma unroll
    for (int n = 0; n < 4; ++n) {
        const size_t co = (size_t)(colBase + n * 16) * K + khalf * 16;
        pZ[n] = WtZ + co;
        pH[n] = WtH + co;
    }

    v8f accZ[4], accH[4];
#pragma unroll
    for (int n = 0; n < 4; ++n) { accZ[n] = {}; accH[n] = {}; }

    // Double-buffered fragments.
    v16h aF[2];
    v16h bzF[2][4], bhF[2][4];

    aF[0] = load_a_frag(pA);
#pragma unroll
    for (int n = 0; n < 4; ++n) {
        bzF[0][n] = *(const v16h*)(pZ[n]);
        bhF[0][n] = *(const v16h*)(pH[n]);
    }

#pragma unroll
    for (int s = 0; s < NSTEP; ++s) {
        const int cur = s & 1;
        const int nxt = cur ^ 1;
        if (s + 1 < NSTEP) {
            // Prefetch next step's fragments (immediate offsets).
            aF[nxt] = load_a_frag(pA + (s + 1) * 32);
#pragma unroll
            for (int n = 0; n < 4; ++n) {
                bzF[nxt][n] = *(const v16h*)(pZ[n] + (s + 1) * 32);
                bhF[nxt][n] = *(const v16h*)(pH[n] + (s + 1) * 32);
            }
        }
#pragma unroll
        for (int n = 0; n < 4; ++n) {
            accZ[n] = __builtin_amdgcn_wmma_f32_16x16x32_f16(
                false, aF[cur], false, bzF[cur][n], (short)0, accZ[n], false, false);
            accH[n] = __builtin_amdgcn_wmma_f32_16x16x32_f16(
                false, aF[cur], false, bhF[cur][n], (short)0, accH[n], false, false);
        }
    }

    // Fused activation epilogue; nontemporal stores keep a/b (268 MB total,
    // dead after the scan) from evicting the L2-resident h buffer.
    const int rowC = mRow + (lane >> 4) * 8;
#pragma unroll
    for (int n = 0; n < 4; ++n) {
        const int col = nColW + n * 16 + (lane & 15);
        const float bzv = bz[col];
        const float bhv = bh[col];
#pragma unroll
        for (int e = 0; e < 8; ++e) {
            const int r  = rowC + e;
            const float gz = accZ[n][e] + bzv;
            const float gh = accH[n][e] + bhv;
            const float kk = fast_sigmoid(gz);
            const float z  = fast_sigmoid(kk);
            const float hw = fast_tanh(fast_sigmoid(gh)) + GRU_EPS;
            const size_t o = (size_t)r * N + col;
            __builtin_nontemporal_store(1.0f - z, a_out + o);
            __builtin_nontemporal_store(z * hw,  b_out + o);
        }
    }
}

// ---------------------------------------------------------------------------
// Chunked scan, pass 1: per-chunk composition.
//   (P, Q) s.t. h_chunk_end = P * h_chunk_start + Q.
// Thread (b, c, hc); consecutive threads in hc -> coalesced.
// B*H*C threads = 524288 -> 16K waves: bandwidth-bound, latency fully hidden.
// ---------------------------------------------------------------------------
__global__ __launch_bounds__(256) void scan_chunk_kernel(
    const float* __restrict__ A, const float* __restrict__ Bv,
    float* __restrict__ chunkP, float* __restrict__ chunkQ,
    int B, int L, int H, int C)
{
    int tid = blockIdx.x * blockDim.x + threadIdx.x;
    if (tid >= B * H * C) return;
    const int hc = tid % H;
    const int c  = (tid / H) % C;
    const int b  = tid / (H * C);
    const int Lc = L / C;
    const size_t base = ((size_t)b * L + (size_t)c * Lc) * H + hc;

    float P = 1.0f, Q = 0.0f;
#pragma unroll 8
    for (int t = 0; t < Lc; ++t) {
        const size_t idx = base + (size_t)t * H;
        const float a  = __builtin_nontemporal_load(A + idx);
        const float bb = __builtin_nontemporal_load(Bv + idx);
        Q = fmaf(a, Q, bb);
        P *= a;
    }
    chunkP[tid] = P;   // layout [b, c, hc]
    chunkQ[tid] = Q;
}

// ---------------------------------------------------------------------------
// Chunked scan, pass 2: sequential combine over C chunks (tiny: 4 MB).
// Writes per-chunk start states (layer 0) and/or the final state (layer 1).
// ---------------------------------------------------------------------------
__global__ __launch_bounds__(256) void scan_combine_kernel(
    const float* __restrict__ chunkP, const float* __restrict__ chunkQ,
    float* __restrict__ hstart,   // [B, C, H] or nullptr
    float* __restrict__ hfinal,   // [B, H]    or nullptr
    int B, int H, int C)
{
    int tid = blockIdx.x * blockDim.x + threadIdx.x;
    if (tid >= B * H) return;
    const int hc = tid % H;
    const int b  = tid / H;

    float h = GRU_EPS;
    for (int c = 0; c < C; ++c) {
        const size_t idx = ((size_t)b * C + c) * H + hc;
        if (hstart) hstart[idx] = h;
        h = fmaf(chunkP[idx], h, chunkQ[idx]);
    }
    if (hfinal) hfinal[tid] = h;
}

// ---------------------------------------------------------------------------
// Chunked scan, pass 3: re-apply per chunk from its start state, writing all
// timesteps (needed only for layer 0; h feeds the layer-1 GEMM from L2).
// ---------------------------------------------------------------------------
__global__ __launch_bounds__(256) void scan_apply_kernel(
    const float* __restrict__ A, const float* __restrict__ Bv,
    const float* __restrict__ hstart,
    float* __restrict__ Hout,
    int B, int L, int H, int C)
{
    int tid = blockIdx.x * blockDim.x + threadIdx.x;
    if (tid >= B * H * C) return;
    const int hc = tid % H;
    const int c  = (tid / H) % C;
    const int b  = tid / (H * C);
    const int Lc = L / C;
    const size_t base = ((size_t)b * L + (size_t)c * Lc) * H + hc;

    float h = hstart[tid];
#pragma unroll 8
    for (int t = 0; t < Lc; ++t) {
        const size_t idx = base + (size_t)t * H;
        const float a  = __builtin_nontemporal_load(A + idx);
        const float bb = __builtin_nontemporal_load(Bv + idx);
        h = fmaf(a, h, bb);
        Hout[idx] = h;   // regular store: h (134 MB) should stay in L2
    }
}

// ---------------------------------------------------------------------------
// Head: out[b] = sum_h hlast[b,h] * Wf[h,0] + bf[0]
// ---------------------------------------------------------------------------
__global__ __launch_bounds__(256) void head_kernel(
    const float* __restrict__ hlast,
    const float* __restrict__ Wf,
    const float* __restrict__ bf,
    float* __restrict__ out, int H)
{
    __shared__ float red[256];
    const int b = blockIdx.x;
    float s = 0.0f;
    for (int i = threadIdx.x; i < H; i += 256)
        s += hlast[(size_t)b * H + i] * Wf[i];
    red[threadIdx.x] = s;
    __syncthreads();
    for (int off = 128; off > 0; off >>= 1) {
        if (threadIdx.x < off) red[threadIdx.x] += red[threadIdx.x + off];
        __syncthreads();
    }
    if (threadIdx.x == 0) out[b] = red[0] + bf[0];
}

// ---------------------------------------------------------------------------
extern "C" void kernel_launch(void* const* d_in, const int* in_sizes, int n_in,
                              void* d_out, int out_size, void* d_ws, size_t ws_size,
                              hipStream_t stream)
{
    (void)in_sizes; (void)n_in; (void)out_size; (void)ws_size;

    const int B = 16, L = 4096, I = 64, H = 512;
    const int M = B * L;    // 65536
    const int C = 64;       // scan chunks (Lc = 64)

    const float* x   = (const float*)d_in[0];
    const float* Wz0 = (const float*)d_in[1];
    const float* bz0 = (const float*)d_in[2];
    const float* Wh0 = (const float*)d_in[3];
    const float* bh0 = (const float*)d_in[4];
    const float* Wz1 = (const float*)d_in[5];
    const float* bz1 = (const float*)d_in[6];
    const float* Wh1 = (const float*)d_in[7];
    const float* bh1 = (const float*)d_in[8];
    const float* Wf  = (const float*)d_in[9];
    const float* bf  = (const float*)d_in[10];
    float* out = (float*)d_out;

    // Workspace carve-up (~410 MB)
    char* ws = (char*)d_ws;
    size_t off = 0;
    auto alloc = [&](size_t bytes) -> void* {
        void* p = ws + off;
        off += (bytes + 255) & ~(size_t)255;
        return p;
    };
    _Float16* WtZ0 = (_Float16*)alloc((size_t)H * I * sizeof(_Float16));
    _Float16* WtH0 = (_Float16*)alloc((size_t)H * I * sizeof(_Float16));
    _Float16* WtZ1 = (_Float16*)alloc((size_t)H * H * sizeof(_Float16));
    _Float16* WtH1 = (_Float16*)alloc((size_t)H * H * sizeof(_Float16));
    float* a_buf  = (float*)alloc((size_t)M * H * sizeof(float));
    float* b_buf  = (float*)alloc((size_t)M * H * sizeof(float));
    float* h_buf  = (float*)alloc((size_t)M * H * sizeof(float));
    float* chunkP = (float*)alloc((size_t)B * C * H * sizeof(float));
    float* chunkQ = (float*)alloc((size_t)B * C * H * sizeof(float));
    float* hstart = (float*)alloc((size_t)B * C * H * sizeof(float));
    float* hlast  = (float*)alloc((size_t)B * H * sizeof(float));

    // 1) Weight transpose + convert (f32 [K,N] -> f16 [N,K])
    {
        int t0 = I * H;
        wt_convert_kernel<<<(t0 + 255) / 256, 256, 0, stream>>>(Wz0, WtZ0, I, H);
        wt_convert_kernel<<<(t0 + 255) / 256, 256, 0, stream>>>(Wh0, WtH0, I, H);
        int t1 = H * H;
        wt_convert_kernel<<<(t1 + 255) / 256, 256, 0, stream>>>(Wz1, WtZ1, H, H);
        wt_convert_kernel<<<(t1 + 255) / 256, 256, 0, stream>>>(Wh1, WtH1, H, H);
    }

    dim3 gemmGrid(M / 64, H / 128);
    const int scanThreads    = B * H * C;   // 524288
    const int combineThreads = B * H;       // 8192

    // 2) Layer 0: dual WMMA GEMM + activations -> (a, b)
    gemm_act_kernel<64><<<gemmGrid, 256, 0, stream>>>(
        x, WtZ0, WtH0, bz0, bh0, a_buf, b_buf, H);

    // 3) Layer 0 chunked scan -> h_buf (all timesteps)
    scan_chunk_kernel<<<(scanThreads + 255) / 256, 256, 0, stream>>>(
        a_buf, b_buf, chunkP, chunkQ, B, L, H, C);
    scan_combine_kernel<<<(combineThreads + 255) / 256, 256, 0, stream>>>(
        chunkP, chunkQ, hstart, nullptr, B, H, C);
    scan_apply_kernel<<<(scanThreads + 255) / 256, 256, 0, stream>>>(
        a_buf, b_buf, hstart, h_buf, B, L, H, C);

    // 4) Layer 1: dual WMMA GEMM + activations (h_buf is L2-resident)
    gemm_act_kernel<512><<<gemmGrid, 256, 0, stream>>>(
        h_buf, WtZ1, WtH1, bz1, bh1, a_buf, b_buf, H);

    // 5) Layer 1 chunked scan: final state only (no apply pass needed)
    scan_chunk_kernel<<<(scanThreads + 255) / 256, 256, 0, stream>>>(
        a_buf, b_buf, chunkP, chunkQ, B, L, H, C);
    scan_combine_kernel<<<(combineThreads + 255) / 256, 256, 0, stream>>>(
        chunkP, chunkQ, nullptr, hlast, B, H, C);

    // 6) Head GEMV
    head_kernel<<<B, 256, 0, stream>>>(hlast, Wf, bf, out, H);
}
